// TransformerBlock_39230231282128
// MI455X (gfx1250) — matrix operand order repaired and checked by reading, not compiled, not run on hardware
//
#include <hip/hip_runtime.h>
#include <hip/hip_bf16.h>

typedef __attribute__((ext_vector_type(16))) __bf16 v16bf;
typedef __attribute__((ext_vector_type(8)))  float  v8f;

// ---------------- problem constants ----------------
#define CB   2
#define CS   2048
#define CD   1024
#define CH   16
#define CHD  64
#define CDFF 4096
#define CBS  (CB*CS)   // 4096 rows

union Frag { v16bf v; uint4 u[2]; };

// ---------------- helpers ----------------
__device__ __forceinline__ float gelu_f(float x) {
    float x3 = x * x * x;
    return 0.5f * x * (1.0f + tanhf(0.79788456f * (x + 0.044715f * x3)));
}

// CDNA5 async global->LDS copy (16B per lane), tracked by ASYNCcnt.
__device__ __forceinline__ void async_ld16B(unsigned int lds, unsigned int goff,
                                            unsigned long long base) {
    asm volatile("global_load_async_to_lds_b128 %0, %1, %2"
                 :: "v"(lds), "v"(goff), "s"(base) : "memory");
}
__device__ __forceinline__ void wait_async0() {
    asm volatile("s_wait_asynccnt 0" ::: "memory");
}

// ---------------- f32 -> bf16 transpose-convert: dst[N][K] = src[K][N] ----------------
__global__ __launch_bounds__(256)
void cvt_tr_f32_bf16(const float* __restrict__ src, __bf16* __restrict__ dst,
                     int K, int N) {
    __shared__ float t[32][33];
    const int k0 = blockIdx.y * 32, n0 = blockIdx.x * 32;
    const int tx = threadIdx.x & 31, ty = threadIdx.x >> 5;   // 32 x 8
#pragma unroll
    for (int i = 0; i < 32; i += 8)
        t[ty + i][tx] = src[(size_t)(k0 + ty + i) * N + n0 + tx];
    __syncthreads();
#pragma unroll
    for (int i = 0; i < 32; i += 8)
        dst[(size_t)(n0 + ty + i) * K + k0 + tx] = (__bf16)t[tx][ty + i];
}

// ---------------- LayerNorm (ddof=1) -> bf16 ----------------
__global__ __launch_bounds__(256)
void layernorm_bf16(const float* __restrict__ x, const float* __restrict__ scale,
                    const float* __restrict__ bias, __bf16* __restrict__ out) {
    __shared__ float sb[256], ssb[256];
    const int row = blockIdx.x;
    const float* xr = x + (size_t)row * CD;
    float v[4], s = 0.f, ss = 0.f;
#pragma unroll
    for (int i = 0; i < 4; ++i) {
        v[i] = xr[threadIdx.x + i * 256];
        s += v[i]; ss += v[i] * v[i];
    }
    sb[threadIdx.x] = s; ssb[threadIdx.x] = ss;
    __syncthreads();
    for (int o = 128; o > 0; o >>= 1) {
        if (threadIdx.x < o) { sb[threadIdx.x] += sb[threadIdx.x+o]; ssb[threadIdx.x] += ssb[threadIdx.x+o]; }
        __syncthreads();
    }
    float mean = sb[0] * (1.0f / CD);
    float var  = (ssb[0] - (float)CD * mean * mean) * (1.0f / (CD - 1));
    float rstd = rsqrtf(var + 1e-5f);
#pragma unroll
    for (int i = 0; i < 4; ++i) {
        int c = threadIdx.x + i * 256;
        out[(size_t)row * CD + c] = (__bf16)(scale[c] * (v[i] - mean) * rstd + bias[c]);
    }
}

// ---------------- WMMA bf16 GEMM: C[M,N] = A[M,K] @ Bt[N,K]^T ----------------
// block tile 64x64, 4 waves, each wave 32x32 (2x2 WMMA accs), BK=32
// A staged row-major, B staged K-major: every fragment = 2x ds_load_b128
// EPI: 0 = write bf16, 1 = f32 + bias + residual, 2 = bf16 gelu(acc+bias)
#define BMT 64
#define BNT 64
#define BKT 32
#define LDA 40   // A row stride (elems): 80B, 16B-aligned
#define LDK 40   // Bt row stride (elems): 80B, 16B-aligned

template<int EPI>
__global__ __launch_bounds__(128)
void gemm_bf16_wmma(const __bf16* __restrict__ A, const __bf16* __restrict__ Bt,
                    const float* __restrict__ bias, const float* __restrict__ res,
                    float* __restrict__ Cf, __bf16* __restrict__ Cb,
                    int M, int N, int K) {
    __shared__ __align__(16) __bf16 As[BMT * LDA];
    __shared__ __align__(16) __bf16 BsT[BNT * LDK];

    const int t    = threadIdx.x;
    const int lane = t & 31;
    const int wave = t >> 5;
    const int wr   = (wave >> 1) * 32;
    const int wc   = (wave & 1) * 32;
    const int half = lane >> 4;
    const int l16  = lane & 15;

    const int m0 = blockIdx.y * BMT;
    const int n0 = blockIdx.x * BNT;

    const unsigned int as_base = (unsigned int)(size_t)(void*)As;
    const unsigned int bs_base = (unsigned int)(size_t)(void*)BsT;
    const unsigned long long a_gbase = (unsigned long long)(size_t)A;
    const unsigned long long b_gbase = (unsigned long long)(size_t)Bt;

    // each thread stages 16 elems of A (one half-row) and 16 elems of Bt (half a column)
    const int arow_ld = t >> 1, aoff_ld = (t & 1) * 16;
    const int bcol_ld = t >> 1, boff_ld = (t & 1) * 16;
    const unsigned int la = as_base + (unsigned int)(arow_ld * LDA + aoff_ld) * 2u;
    const unsigned int lb = bs_base + (unsigned int)(bcol_ld * LDK + boff_ld) * 2u;

    v8f acc[2][2] = {};

    for (int k0 = 0; k0 < K; k0 += BKT) {
        unsigned int ga = (unsigned int)(((size_t)(m0 + arow_ld) * K + k0 + aoff_ld) * 2);
        unsigned int gb = (unsigned int)(((size_t)(n0 + bcol_ld) * K + k0 + boff_ld) * 2);
        async_ld16B(la,      ga,      a_gbase);
        async_ld16B(la + 16, ga + 16, a_gbase);
        async_ld16B(lb,      gb,      b_gbase);
        async_ld16B(lb + 16, gb + 16, b_gbase);
        wait_async0();
        __syncthreads();

        Frag af[2], bf[2];
#pragma unroll
        for (int f = 0; f < 2; ++f) {
            const __bf16* ap = &As[(wr + f * 16 + l16) * LDA];
            af[f].u[0] = *(const uint4*)(ap + half * 8);
            af[f].u[1] = *(const uint4*)(ap + 16 + half * 8);
            const __bf16* bp = &BsT[(wc + f * 16 + l16) * LDK];
            bf[f].u[0] = *(const uint4*)(bp + half * 8);
            bf[f].u[1] = *(const uint4*)(bp + 16 + half * 8);
        }
#pragma unroll
        for (int i = 0; i < 2; ++i)
#pragma unroll
            for (int j = 0; j < 2; ++j)
                acc[i][j] = __builtin_amdgcn_wmma_f32_16x16x32_bf16(
                    false, af[i].v, false, bf[j].v, (short)0, acc[i][j], false, false);
        __syncthreads();
    }

#pragma unroll
    for (int i = 0; i < 2; ++i) {
#pragma unroll
        for (int j = 0; j < 2; ++j) {
            int col = n0 + wc + j * 16 + l16;
#pragma unroll
            for (int r = 0; r < 8; ++r) {
                int row = m0 + wr + i * 16 + half * 8 + r;     // ISA C/D layout
                float v = acc[i][j][r];
                size_t idx = (size_t)row * N + col;
                if (EPI == 0) {
                    Cb[idx] = (__bf16)v;
                } else if (EPI == 1) {
                    Cf[idx] = v + bias[col] + res[idx];
                } else {
                    Cb[idx] = (__bf16)gelu_f(v + bias[col]);
                }
            }
        }
    }
}

// ---------------- flash attention (causal) ----------------
// grid: (S/64, H, B), block 128. Block cooperatively stages 32-key K/V tiles in
// LDS (K via async b128, V transposed to [hd][key]); each wave owns a 16-query
// tile with online softmax; P goes through LDS for the C->A layout transpose.
#define LDKS 72  // Ks row stride (elems): 144B, 16B-aligned
#define LDVS 40  // Vs row stride (elems): 80B, 16B-aligned

__global__ __launch_bounds__(128)
void attn_fwd(const __bf16* __restrict__ Q, const __bf16* __restrict__ Km,
              const __bf16* __restrict__ V, __bf16* __restrict__ ctx) {
    const int b = blockIdx.z, h = blockIdx.y;
    const int t = threadIdx.x;
    const int wave = t >> 5, lane = t & 31;
    const int half = lane >> 4, l16 = lane & 15;
    const int qblk = blockIdx.x * 64;
    const int q0 = qblk + wave * 16;

    const __bf16* qp = Q  + ((size_t)b * CS) * CD + (size_t)h * CHD;
    const __bf16* kp = Km + ((size_t)b * CS) * CD + (size_t)h * CHD;
    const __bf16* vp = V  + ((size_t)b * CS) * CD + (size_t)h * CHD;

    __shared__ __align__(16) __bf16 Ks[32 * LDKS];       // [key][hd]
    __shared__ __align__(16) __bf16 Vs[64 * LDVS];       // [hd][key]
    __shared__ __align__(16) __bf16 probs[4][16 * LDVS]; // per-wave [qrow][key]

    // Q fragments: 16 rows x 64 hd = two K32 A-frags (2x b128 global each)
    Frag qf[2];
    {
        const __bf16* qrp = qp + (size_t)(q0 + l16) * CD;
#pragma unroll
        for (int f = 0; f < 2; ++f) {
            qf[f].u[0] = *(const uint4*)(qrp + f * 32 + half * 8);
            qf[f].u[1] = *(const uint4*)(qrp + f * 32 + 16 + half * 8);
        }
    }

    v8f oacc[4] = {};
    float mrow[8], lrow[8];
#pragma unroll
    for (int r = 0; r < 8; ++r) { mrow[r] = -1e30f; lrow[r] = 0.f; }

    const unsigned int ks_base = (unsigned int)(size_t)(void*)Ks;
    const unsigned long long k_gbase = (unsigned long long)(size_t)kp;
    const int key_ld = t >> 2, hdoff_ld = (t & 3) * 16;   // staging assignment
    const unsigned int lk = ks_base + (unsigned int)(key_ld * LDKS + hdoff_ld) * 2u;

    const int kbmax = (qblk + 63) >> 5;   // last 32-key block for this query block
    for (int kb = 0; kb <= kbmax; ++kb) {
        const int kcol = kb * 32;
        // ---- cooperative staging ----
        unsigned int gk = (unsigned int)(((size_t)(kcol + key_ld) * CD + hdoff_ld) * 2);
        async_ld16B(lk,      gk,      k_gbase);
        async_ld16B(lk + 16, gk + 16, k_gbase);
        {   // V: registers -> transposed LDS [hd][key]
            const uint4* vsrc = (const uint4*)(vp + (size_t)(kcol + key_ld) * CD + hdoff_ld);
            uint4 d0 = vsrc[0], d1 = vsrc[1];
            union { uint4 u; __bf16 e[8]; } c0, c1;
            c0.u = d0; c1.u = d1;
#pragma unroll
            for (int e = 0; e < 8; ++e) {
                Vs[(hdoff_ld + e) * LDVS + key_ld]     = c0.e[e];
                Vs[(hdoff_ld + 8 + e) * LDVS + key_ld] = c1.e[e];
            }
        }
        wait_async0();
        __syncthreads();

        if (kcol <= q0 + 15) {   // this wave's tile intersects the key block
            // ---- scores: Q(16x64) @ K^T(64x32) ----
            v8f sc[2];
#pragma unroll
            for (int nt = 0; nt < 2; ++nt) {
                const __bf16* krow = &Ks[(nt * 16 + l16) * LDKS];
                Frag k0f, k1f;
                k0f.u[0] = *(const uint4*)(krow + half * 8);
                k0f.u[1] = *(const uint4*)(krow + 16 + half * 8);
                k1f.u[0] = *(const uint4*)(krow + 32 + half * 8);
                k1f.u[1] = *(const uint4*)(krow + 32 + 16 + half * 8);
                v8f z = {};
                z = __builtin_amdgcn_wmma_f32_16x16x32_bf16(false, qf[0].v, false, k0f.v, (short)0, z, false, false);
                sc[nt] = __builtin_amdgcn_wmma_f32_16x16x32_bf16(false, qf[1].v, false, k1f.v, (short)0, z, false, false);
            }
            // ---- scale + causal mask ----
#pragma unroll
            for (int nt = 0; nt < 2; ++nt) {
                int key = kcol + nt * 16 + l16;
#pragma unroll
                for (int r = 0; r < 8; ++r) {
                    int qrow = q0 + half * 8 + r;
                    float v = sc[nt][r] * 0.125f;   // 1/sqrt(64)
                    sc[nt][r] = (key <= qrow) ? v : -1e30f;
                }
            }
            // ---- online softmax ----
#pragma unroll
            for (int r = 0; r < 8; ++r) {
                float mx = fmaxf(sc[0][r], sc[1][r]);
#pragma unroll
                for (int o = 1; o < 16; o <<= 1) mx = fmaxf(mx, __shfl_xor(mx, o, 32));
                float mnew = fmaxf(mrow[r], mx);
                float alpha = expf(mrow[r] - mnew);
                mrow[r] = mnew;
                float p0 = expf(sc[0][r] - mnew);
                float p1 = expf(sc[1][r] - mnew);
                sc[0][r] = p0; sc[1][r] = p1;
                float rs = p0 + p1;
#pragma unroll
                for (int o = 1; o < 16; o <<= 1) rs += __shfl_xor(rs, o, 32);
                lrow[r] = lrow[r] * alpha + rs;
#pragma unroll
                for (int c = 0; c < 4; ++c) oacc[c][r] *= alpha;
            }
            // ---- C-layout probs -> LDS -> A-layout frag (intra-wave transpose) ----
            __bf16* pw = probs[wave];
#pragma unroll
            for (int nt = 0; nt < 2; ++nt)
#pragma unroll
                for (int r = 0; r < 8; ++r)
                    pw[(half * 8 + r) * LDVS + nt * 16 + l16] = (__bf16)sc[nt][r];
            asm volatile("s_wait_dscnt 0" ::: "memory");
            Frag pf;
            pf.u[0] = *(const uint4*)(pw + l16 * LDVS + half * 8);
            pf.u[1] = *(const uint4*)(pw + l16 * LDVS + 16 + half * 8);
            // ---- P(16x32) @ V(32x64) ----
#pragma unroll
            for (int c = 0; c < 4; ++c) {
                const __bf16* vrow = &Vs[(c * 16 + l16) * LDVS];
                Frag vf;
                vf.u[0] = *(const uint4*)(vrow + half * 8);
                vf.u[1] = *(const uint4*)(vrow + 16 + half * 8);
                oacc[c] = __builtin_amdgcn_wmma_f32_16x16x32_bf16(false, pf.v, false, vf.v, (short)0, oacc[c], false, false);
            }
        }
        __syncthreads();
    }

    // normalize and write ctx (already in [B,S,H*HD] layout)
#pragma unroll
    for (int c = 0; c < 4; ++c)
#pragma unroll
        for (int r = 0; r < 8; ++r) {
            int row = q0 + half * 8 + r;
            ctx[((size_t)b * CS + row) * CD + h * CHD + c * 16 + l16] =
                (__bf16)(oacc[c][r] / lrow[r]);
        }
}

// ---------------- launch ----------------
extern "C" void kernel_launch(void* const* d_in, const int* in_sizes, int n_in,
                              void* d_out, int out_size, void* d_ws, size_t ws_size,
                              hipStream_t stream) {
    const float* x    = (const float*)d_in[0];
    const float* Wq   = (const float*)d_in[1];
    const float* Wk   = (const float*)d_in[2];
    const float* Wv   = (const float*)d_in[3];
    const float* Wo   = (const float*)d_in[4];
    const float* bo   = (const float*)d_in[5];
    const float* W1   = (const float*)d_in[6];
    const float* b1   = (const float*)d_in[7];
    const float* W2   = (const float*)d_in[8];
    const float* b2   = (const float*)d_in[9];
    const float* ln1s = (const float*)d_in[10];
    const float* ln1b = (const float*)d_in[11];
    const float* ln2s = (const float*)d_in[12];
    const float* ln2b = (const float*)d_in[13];
    float* out = (float*)d_out;

    // workspace carve-up
    char* p = (char*)d_ws;
    auto carve = [&](size_t bytes) { char* r = p; p += (bytes + 255) & ~(size_t)255; return r; };
    __bf16* wq_t  = (__bf16*)carve((size_t)CD * CD * 2);     // [N][K]
    __bf16* wk_t  = (__bf16*)carve((size_t)CD * CD * 2);
    __bf16* wv_t  = (__bf16*)carve((size_t)CD * CD * 2);
    __bf16* wo_t  = (__bf16*)carve((size_t)CD * CD * 2);
    __bf16* w1_t  = (__bf16*)carve((size_t)CD * CDFF * 2);   // [4D][D]
    __bf16* w2_t  = (__bf16*)carve((size_t)CDFF * CD * 2);   // [D][4D]
    __bf16* h_b   = (__bf16*)carve((size_t)CBS * CD * 2);
    __bf16* q_b   = (__bf16*)carve((size_t)CBS * CD * 2);
    __bf16* k_b   = (__bf16*)carve((size_t)CBS * CD * 2);
    __bf16* v_b   = (__bf16*)carve((size_t)CBS * CD * 2);
    __bf16* ctx_b = (__bf16*)carve((size_t)CBS * CD * 2);
    float*  x1_f  = (float*) carve((size_t)CBS * CD * 4);
    __bf16* h2_b  = (__bf16*)carve((size_t)CBS * CD * 2);
    __bf16* u_b   = (__bf16*)carve((size_t)CBS * CDFF * 2);

    // weight transpose-convert: dst[N][K] = src[K][N], bf16
    dim3 gDD(CD / 32, CD / 32);
    cvt_tr_f32_bf16<<<gDD, 256, 0, stream>>>(Wq, wq_t, CD, CD);
    cvt_tr_f32_bf16<<<gDD, 256, 0, stream>>>(Wk, wk_t, CD, CD);
    cvt_tr_f32_bf16<<<gDD, 256, 0, stream>>>(Wv, wv_t, CD, CD);
    cvt_tr_f32_bf16<<<gDD, 256, 0, stream>>>(Wo, wo_t, CD, CD);
    cvt_tr_f32_bf16<<<dim3(CDFF / 32, CD / 32), 256, 0, stream>>>(W1, w1_t, CD, CDFF);
    cvt_tr_f32_bf16<<<dim3(CD / 32, CDFF / 32), 256, 0, stream>>>(W2, w2_t, CDFF, CD);

    // LN1
    layernorm_bf16<<<CBS, 256, 0, stream>>>(x, ln1s, ln1b, h_b);

    // QKV projections
    dim3 gQKV(CD / BNT, CBS / BMT);
    gemm_bf16_wmma<0><<<gQKV, 128, 0, stream>>>(h_b, wq_t, nullptr, nullptr, nullptr, q_b, CBS, CD, CD);
    gemm_bf16_wmma<0><<<gQKV, 128, 0, stream>>>(h_b, wk_t, nullptr, nullptr, nullptr, k_b, CBS, CD, CD);
    gemm_bf16_wmma<0><<<gQKV, 128, 0, stream>>>(h_b, wv_t, nullptr, nullptr, nullptr, v_b, CBS, CD, CD);

    // attention
    attn_fwd<<<dim3(CS / 64, CH, CB), 128, 0, stream>>>(q_b, k_b, v_b, ctx_b);

    // output projection + bias + residual -> x1
    gemm_bf16_wmma<1><<<gQKV, 128, 0, stream>>>(ctx_b, wo_t, bo, x, x1_f, nullptr, CBS, CD, CD);

    // LN2
    layernorm_bf16<<<CBS, 256, 0, stream>>>(x1_f, ln2s, ln2b, h2_b);

    // FFN
    dim3 gF1(CDFF / BNT, CBS / BMT);
    gemm_bf16_wmma<2><<<gF1, 128, 0, stream>>>(h2_b, w1_t, b1, nullptr, nullptr, u_b, CBS, CDFF, CD);
    gemm_bf16_wmma<1><<<gQKV, 128, 0, stream>>>(u_b, w2_t, b2, x1_f, out, nullptr, CBS, CD, CDFF);

    (void)in_sizes; (void)n_in; (void)out_size; (void)ws_size;
}
